// AttentionHead_3702261809374
// MI455X (gfx1250) — matrix operand order repaired and checked by reading, not compiled, not run on hardware
//
#include <hip/hip_runtime.h>

// ---------------------------------------------------------------------------
// AttentionHead for MI455X (gfx1250, wave32, WMMA bf16 16x16x32)
//  wprep : W[C,H] f32 -> WtT[3][H][C] bf16 (transposed, L2-resident)
//  proj  : q,k,v f32 [B,T,C] x WtT -> qh,kh bf16 [B,T,H], vt bf16 [B,H,T]
//          (pure register GEMM: A from global f32+cvt, B contiguous global)
//  fattn : per-wave flash attention, 16 query rows x 32-key blocks,
//          K/V double-buffered into LDS via global_load_async_to_lds_b128.
// ---------------------------------------------------------------------------

typedef __attribute__((ext_vector_type(16))) __bf16 v16bf;
typedef __attribute__((ext_vector_type(8)))  float  v8f;
typedef __attribute__((ext_vector_type(4)))  int    i32x4v;

#define Bn 4
#define Tn 4096
#define Cn 1024
#define Hn 64

#if defined(__HIP_DEVICE_COMPILE__) && \
    __has_builtin(__builtin_amdgcn_global_load_async_to_lds_b128)
#define HAVE_ASYNC_LDS 1
#else
#define HAVE_ASYNC_LDS 0
#endif

union BV16 { uint4 u[2]; v16bf v; };

__device__ __forceinline__ __bf16 f2bf(float f) { return (__bf16)f; }

__device__ __forceinline__ v8f wmma_bf16(v16bf a, v16bf b, v8f c) {
  return __builtin_amdgcn_wmma_f32_16x16x32_bf16(false, a, false, b,
                                                 (short)0, c, false, false);
}

// 32 contiguous bytes (16 bf16) -> B operand fragment (global or LDS)
__device__ __forceinline__ v16bf load_v16bf(const __bf16* p) {
  BV16 r;
  r.u[0] = *(const uint4*)(p);
  r.u[1] = *(const uint4*)(p + 8);
  return r.v;
}

// A operand from a bf16 row: cols [c0+8h..+7] and [c0+16+8h..+7]
__device__ __forceinline__ v16bf load_a_bf16(const __bf16* row, int c0, int hi) {
  BV16 r;
  r.u[0] = *(const uint4*)(row + c0 + 8 * hi);
  r.u[1] = *(const uint4*)(row + c0 + 16 + 8 * hi);
  return r.v;
}

// A operand from an f32 row, converting to bf16
__device__ __forceinline__ v16bf load_a_f32(const float* rowc, int hi) {
  const float4* p0 = (const float4*)(rowc + 8 * hi);
  const float4* p1 = (const float4*)(rowc + 16 + 8 * hi);
  float4 f0 = p0[0], f1 = p0[1], f2 = p1[0], f3 = p1[1];
  v16bf a;
  a[0] = f2bf(f0.x); a[1] = f2bf(f0.y); a[2]  = f2bf(f0.z); a[3]  = f2bf(f0.w);
  a[4] = f2bf(f1.x); a[5] = f2bf(f1.y); a[6]  = f2bf(f1.z); a[7]  = f2bf(f1.w);
  a[8] = f2bf(f2.x); a[9] = f2bf(f2.y); a[10] = f2bf(f2.z); a[11] = f2bf(f2.w);
  a[12] = f2bf(f3.x); a[13] = f2bf(f3.y); a[14] = f2bf(f3.z); a[15] = f2bf(f3.w);
  return a;
}

// 16-byte global -> LDS copy: async on gfx1250 (ASYNCcnt-tracked DMA path)
__device__ __forceinline__ void copy16_g2l(const __bf16* g, __bf16* l) {
#if HAVE_ASYNC_LDS
  __builtin_amdgcn_global_load_async_to_lds_b128(
      (__attribute__((address_space(1))) i32x4v*)(
          __attribute__((address_space(1))) void*)(const void*)g,
      (__attribute__((address_space(3))) i32x4v*)(
          __attribute__((address_space(3))) void*)(void*)l,
      0, 0);
#else
  *(uint4*)l = *(const uint4*)g;
#endif
}

__device__ __forceinline__ void wait_async_le16() {
#if HAVE_ASYNC_LDS
#if __has_builtin(__builtin_amdgcn_s_wait_asynccnt)
  __builtin_amdgcn_s_wait_asynccnt(16);
#else
  asm volatile("s_wait_asynccnt 0x10" ::: "memory");
#endif
#endif
  __builtin_amdgcn_wave_barrier();
}

__device__ __forceinline__ void wait_async_le0() {
#if HAVE_ASYNC_LDS
#if __has_builtin(__builtin_amdgcn_s_wait_asynccnt)
  __builtin_amdgcn_s_wait_asynccnt(0);
#else
  asm volatile("s_wait_asynccnt 0x0" ::: "memory");
#endif
#endif
  __builtin_amdgcn_wave_barrier();
}

__device__ __forceinline__ void wave_lds_fence() {
  __builtin_amdgcn_wave_barrier();
  asm volatile("s_wait_dscnt 0x0" ::: "memory");
  __builtin_amdgcn_wave_barrier();
}

// ---------------------------------------------------------------------------
// Prep: WtT[proj][h][c] = bf16(W[c][h]); 196608 elements total.
// ---------------------------------------------------------------------------
__global__ __launch_bounds__(256) void wprep_kernel(
    const float* __restrict__ Wq, const float* __restrict__ Wk,
    const float* __restrict__ Wv, __bf16* __restrict__ wtT) {
  int e    = blockIdx.x * 256 + threadIdx.x;  // 0..196607
  int proj = e >> 16;
  int rem  = e & 65535;
  int h    = rem >> 10;
  int c    = rem & 1023;
  const float* W = (proj == 0) ? Wq : ((proj == 1) ? Wk : Wv);
  wtT[e] = f2bf(W[(size_t)c * Hn + h]);
}

// ---------------------------------------------------------------------------
// Projections. grid = (Tn/128, 3, Bn), block = 256 (8 waves).
// Each wave: 16x64 output tile, K loop over C in 32-chunks, no LDS/barriers.
// ---------------------------------------------------------------------------
__global__ __launch_bounds__(256) void proj_kernel(
    const float* __restrict__ q, const float* __restrict__ k,
    const float* __restrict__ v, const __bf16* __restrict__ wtT,
    __bf16* __restrict__ qh, __bf16* __restrict__ kh,
    __bf16* __restrict__ vt) {
  const int tid  = threadIdx.x;
  const int wave = tid >> 5;
  const int lane = tid & 31;
  const int n    = lane & 15;
  const int hi   = lane >> 4;
  const int b    = blockIdx.z;
  const int proj = blockIdx.y;
  const int t0   = (blockIdx.x * 8 + wave) * 16;

  const float* X = (proj == 0) ? q : ((proj == 1) ? k : v);
  const __bf16* wbase = wtT + (size_t)proj * Hn * Cn;   // [64][1024]
  const float* xrow = X + ((size_t)b * Tn + (t0 + n)) * Cn;

  v8f acc[4];
#pragma unroll
  for (int i = 0; i < 4; i++)
    acc[i] = (v8f){0.f, 0.f, 0.f, 0.f, 0.f, 0.f, 0.f, 0.f};

  for (int c0 = 0; c0 < Cn; c0 += 32) {
    __builtin_prefetch(xrow + c0 + 128, 0, 1);  // stream X ahead
    v16bf a = load_a_f32(xrow + c0, hi);
#pragma unroll
    for (int nt = 0; nt < 4; nt++) {
      v16bf bb = load_a_bf16(wbase + (size_t)(nt * 16 + n) * Cn, c0, hi);
      acc[nt] = wmma_bf16(a, bb, acc[nt]);
    }
  }

  // Branch-free strided epilogue: (rs,cs) = (H,1) row-major / (1,T) for V^T
  __bf16* dst;
  size_t rs, cs;
  if (proj == 0)      { dst = qh + (size_t)b * Tn * Hn; rs = Hn; cs = 1;  }
  else if (proj == 1) { dst = kh + (size_t)b * Tn * Hn; rs = Hn; cs = 1;  }
  else                { dst = vt + (size_t)b * Hn * Tn; rs = 1;  cs = Tn; }
#pragma unroll
  for (int nt = 0; nt < 4; nt++) {
#pragma unroll
    for (int r = 0; r < 8; r++) {
      size_t row = (size_t)(t0 + r + 8 * hi);
      size_t col = (size_t)(nt * 16 + n);
      dst[row * rs + col * cs] = f2bf(acc[nt][r]);
    }
  }
}

// ---------------------------------------------------------------------------
// Flash attention. grid = (Tn/64, 1, Bn), block = 128 (4 waves).
// Per-wave 16-row query tile; 32-key blocks double-buffered in LDS
// (16 async b128 copies per block: 4KB K slab + 64x64B V slabs).
// ---------------------------------------------------------------------------
__global__ __launch_bounds__(128) void fattn_kernel(
    const __bf16* __restrict__ qh, const __bf16* __restrict__ kh,
    const __bf16* __restrict__ vt, float* __restrict__ out) {
  __shared__ __align__(16) __bf16 Kbuf[4][2][32 * 64];  // 32KB
  __shared__ __align__(16) __bf16 Vbuf[4][2][64 * 32];  // 32KB
  __shared__ __align__(16) __bf16 Plds[4][16 * 32];     // 4KB

  const int tid  = threadIdx.x;
  const int wave = tid >> 5;
  const int lane = tid & 31;
  const int n    = lane & 15;
  const int hi   = lane >> 4;
  const int b    = blockIdx.z;
  const int t0   = (blockIdx.x * 4 + wave) * 16;
  const int thi  = t0 + 15;

  const __bf16* kbase = kh + (size_t)b * Tn * Hn;
  const __bf16* vbase = vt + (size_t)b * Hn * Tn;
  __bf16* myP = &Plds[wave][0];

  const __bf16* qrow = qh + ((size_t)b * Tn + t0 + n) * Hn;
  v16bf aq0 = load_v16bf(qrow + 16 * hi);
  v16bf aq1 = load_v16bf(qrow + 32 + 16 * hi);

  v8f o[4];
#pragma unroll
  for (int i = 0; i < 4; i++)
    o[i] = (v8f){0.f, 0.f, 0.f, 0.f, 0.f, 0.f, 0.f, 0.f};
  float ms[8], ls[8];
#pragma unroll
  for (int r = 0; r < 8; r++) { ms[r] = -1e30f; ls[r] = 0.f; }

  const float inv_scale = 0.125f;  // 1/sqrt(H)
  const int nblocks = (thi >> 5) + 1;

  // issue fill of 32-key block `s0` into buffer `buf` (16 x 16B per lane)
  auto fill_kv = [&](int buf, int s0) {
    const __bf16* ksrc = kbase + (size_t)s0 * Hn;  // contiguous 4KB slab
#pragma unroll
    for (int i = 0; i < 8; i++) {
      int off = (lane + 32 * i) * 8;  // bf16 elements, 16B chunks
      copy16_g2l(ksrc + off, &Kbuf[wave][buf][off]);
    }
#pragma unroll
    for (int i = 0; i < 8; i++) {
      int c   = lane + 32 * i;    // 256 chunks of 16B
      int row = c >> 2;           // 0..63 (head dim)
      int sub = c & 3;            // 16B piece within 64B row segment
      copy16_g2l(vbase + (size_t)row * Tn + s0 + sub * 8,
                 &Vbuf[wave][buf][row * 32 + sub * 8]);
    }
  };

  fill_kv(0, 0);  // prologue

  for (int ib = 0; ib < nblocks; ib++) {
    const int s0  = ib << 5;
    const int cur = ib & 1;
    const bool more = (ib + 1) < nblocks;
    if (more) {
      fill_kv(cur ^ 1, s0 + 32);  // overlap next block's copy with compute
      if (s0 + 64 <= thi)         // light L2 prefetch two blocks out
        __builtin_prefetch(kbase + (size_t)(s0 + 64) * Hn + lane * 64, 0, 1);
      wait_async_le16();          // current buffer's 16 copies are done
    } else {
      wait_async_le0();
    }

    // ---- S = Q K^T over keys [s0, s0+31] (LDS-resident K) ----------------
    v8f c0 = (v8f){0.f, 0.f, 0.f, 0.f, 0.f, 0.f, 0.f, 0.f};
    v8f c1 = c0;
    {
      const __bf16* k0 = &Kbuf[wave][cur][(n) * 64];
      const __bf16* k1 = &Kbuf[wave][cur][(n + 16) * 64];
      v16bf b00 = load_v16bf(k0 + 16 * hi);
      v16bf b01 = load_v16bf(k0 + 32 + 16 * hi);
      v16bf b10 = load_v16bf(k1 + 16 * hi);
      v16bf b11 = load_v16bf(k1 + 32 + 16 * hi);
      c0 = wmma_bf16(aq0, b00, c0);
      c0 = wmma_bf16(aq1, b01, c0);
      c1 = wmma_bf16(aq0, b10, c1);
      c1 = wmma_bf16(aq1, b11, c1);
    }

    // ---- online softmax (rows r+8*hi; 16-lane shfl reductions) -----------
    float alpha[8], p0v[8], p1v[8];
#pragma unroll
    for (int r = 0; r < 8; r++) {
      const int t = t0 + r + 8 * hi;
      float x0 = (s0 + n      <= t) ? c0[r] * inv_scale : -1e30f;
      float x1 = (s0 + 16 + n <= t) ? c1[r] * inv_scale : -1e30f;
      float mx = fmaxf(x0, x1);
#pragma unroll
      for (int s = 1; s < 16; s <<= 1) mx = fmaxf(mx, __shfl_xor(mx, s, 32));
      float mnew = fmaxf(ms[r], mx);
      float e0 = __expf(x0 - mnew);
      float e1 = __expf(x1 - mnew);
      float a  = __expf(ms[r] - mnew);
      float rsum = e0 + e1;
#pragma unroll
      for (int s = 1; s < 16; s <<= 1) rsum += __shfl_xor(rsum, s, 32);
      ls[r]    = ls[r] * a + rsum;
      ms[r]    = mnew;
      alpha[r] = a;
      p0v[r]   = e0;
      p1v[r]   = e1;
    }
#pragma unroll
    for (int r = 0; r < 8; r++) {
      o[0][r] *= alpha[r]; o[1][r] *= alpha[r];
      o[2][r] *= alpha[r]; o[3][r] *= alpha[r];
    }

    // ---- re-layout P (C layout -> A layout) via per-wave LDS -------------
#pragma unroll
    for (int r = 0; r < 8; r++) {
      myP[(r + 8 * hi) * 32 + n]      = f2bf(p0v[r]);
      myP[(r + 8 * hi) * 32 + 16 + n] = f2bf(p1v[r]);
    }
    wave_lds_fence();

    BV16 ap;
    ap.u[0] = *(const uint4*)(myP + n * 32 + 16 * hi);
    ap.u[1] = *(const uint4*)(myP + n * 32 + 16 * hi + 8);

    // ---- O += P * V (V slab LDS-resident, contiguous 32B per lane) -------
#pragma unroll
    for (int nt = 0; nt < 4; nt++) {
      v16bf bv = load_v16bf(&Vbuf[wave][cur][(nt * 16 + n) * 32 + 16 * hi]);
      o[nt] = wmma_bf16(ap.v, bv, o[nt]);
    }
  }

  // ---- normalize and store (16 lanes -> 64B coalesced runs) --------------
#pragma unroll
  for (int nt = 0; nt < 4; nt++) {
#pragma unroll
    for (int r = 0; r < 8; r++) {
      int row = t0 + r + 8 * hi;
      out[((size_t)b * Tn + row) * Hn + nt * 16 + n] = o[nt][r] / ls[r];
    }
  }
}

// ---------------------------------------------------------------------------
extern "C" void kernel_launch(void* const* d_in, const int* in_sizes, int n_in,
                              void* d_out, int out_size, void* d_ws,
                              size_t ws_size, hipStream_t stream) {
  (void)in_sizes; (void)n_in; (void)out_size; (void)ws_size;
  const float* q  = (const float*)d_in[0];
  const float* k  = (const float*)d_in[1];
  const float* v  = (const float*)d_in[2];
  const float* Wq = (const float*)d_in[3];
  const float* Wk = (const float*)d_in[4];
  const float* Wv = (const float*)d_in[5];
  // d_in[6] = mask: causal, handled statically in-kernel.

  __bf16* qh  = (__bf16*)d_ws;
  __bf16* kh  = qh + (size_t)Bn * Tn * Hn;
  __bf16* vt  = kh + (size_t)Bn * Tn * Hn;
  __bf16* wtT = vt + (size_t)Bn * Tn * Hn;  // 3*64*1024; total ~6.4MB

  wprep_kernel<<<dim3(768), 256, 0, stream>>>(Wq, Wk, Wv, wtT);

  dim3 g1(Tn / 128, 3, Bn);  // (32, 3, 4)
  proj_kernel<<<g1, 256, 0, stream>>>(q, k, v, wtT, qh, kh, vt);

  dim3 g2(Tn / 64, 1, Bn);   // (64, 1, 4)
  fattn_kernel<<<g2, 128, 0, stream>>>(qh, kh, vt, (float*)d_out);
}
